// DocREModel_9053791060393
// MI455X (gfx1250) — compile-verified, matrix operand order checked by reading, not run on hardware
//
#include <hip/hip_runtime.h>
#include <hip/hip_bf16.h>

// ---------------------------------------------------------------------------
// DocRE forward pipeline for gfx1250 (MI455X).
// Heavy GEMMs (head/tail projection 156 GF, bilinear 59 GF, fmap 11 GF,
// scores MLP) run on v_wmma_f32_16x16x32_f16 with LDS-staged 64x64 block
// tiles (8 waves/block, 2 WMMA subtiles per wave per K-step).  Edge tiles are
// zero-padded at the global->LDS staging step so the WMMA inner loop is
// completely branch-free.  Everything else (gathers, softmax over 8 mentions,
// h=12 pair overlap, conv3x3, normalizations) is VALU.
// ---------------------------------------------------------------------------

typedef __attribute__((ext_vector_type(16))) _Float16 v16h;
typedef __attribute__((ext_vector_type(8)))  _Float16 v8h_t;
typedef __attribute__((ext_vector_type(8)))  float    v8f;

// Problem constants
#define BS 4
#define LL 1024
#define DD 768
#define HH 12
#define NE 42
#define NM 8
#define KK 97
#define PP 128
#define FF 256
#define RR 256
#define NPAIR (BS * PP)          // 512
#define NROW  (NPAIR * KK)       // 49664
#define CAT   (DD + FF)          // 1024

// GEMM block tile
#define BM 64
#define BN 64
#define BK 32

// ---------------------------------------------------------------------------
// Block-tiled WMMA GEMM:  C[M,N] = act(A[M,Kd] * Bcm + bias)
//   A   : f16 row-major, arbitrary row stride (supports strided hs[n*97+k] view)
//   Bcm : f16 column-major ([N][Kd])
//   act : 0 -> f32 C, 1 -> tanh -> f16 C, 2 -> fused bilinear epilogue
//         (logits[m, bz] += sum_n C[m,n] * ts[m, bz, n], via atomics)
// 8 waves/block; wave w owns m-subtile (w>>1) and n-subtiles 2*(w&1)..+1.
// Fragment layout per CDNA5 ISA 7.12.2: lanes 0-15 = row/col (lane&15) with
// k = [0..7,16..23]; lanes 16-31 use k = [8..15,24..31].
// ---------------------------------------------------------------------------
__global__ __launch_bounds__(256)
void wmma_gemm_block(const _Float16* __restrict__ A, long aRowStride, long aBatch,
                     const _Float16* __restrict__ Bcm, long bBatch,
                     const float* __restrict__ bias,
                     float* __restrict__ Cf, _Float16* __restrict__ Ch, long cBatch,
                     const _Float16* __restrict__ tsFused,
                     float* __restrict__ logitsFused,
                     int M, int N, int Kd, int act) {
  __shared__ __align__(16) _Float16 As[BM][BK];
  __shared__ __align__(16) _Float16 Bs[BN][BK];

  int tid  = threadIdx.x;
  int lane = tid & 31;
  int w    = tid >> 5;
  int n0   = blockIdx.x * BN;
  int m0   = blockIdx.y * BM;
  int bz   = blockIdx.z;
  const _Float16* Ab = A   + (long)bz * aBatch;
  const _Float16* Bb = Bcm + (long)bz * bBatch;

  int sm  = w >> 1;          // m-subtile 0..3
  int sn0 = (w & 1) * 2;     // first of two n-subtiles

  int ldr = tid >> 2;        // 0..63: staged row (A) / col (B)
  int ldk = (tid & 3) * 8;   // 0,8,16,24: k offset of the 16B chunk

  int fr  = (lane & 15);     // fragment row/col within subtile
  int kb  = (lane >> 4) * 8; // fragment k base (0 or 8)

  v8f acc0 = {}, acc1 = {};

  for (int kt = 0; kt < Kd; kt += BK) {
    // ---- stage tiles to LDS (zero-pad ragged edges once, here) ----
    {
      v8h_t va = {};
      int r = m0 + ldr;
      if (r < M) va = *(const v8h_t*)(Ab + (long)r * aRowStride + kt + ldk);
      *(v8h_t*)&As[ldr][ldk] = va;
      v8h_t vb = {};
      int c = n0 + ldr;
      if (c < N) vb = *(const v8h_t*)(Bb + (long)c * Kd + kt + ldk);
      *(v8h_t*)&Bs[ldr][ldk] = vb;
    }
    __syncthreads();

    // ---- prefetch next K chunk into L2 while we compute ----
    if (kt + BK < Kd) {
      int r = m0 + ldr;
      if (r < M) __builtin_prefetch(Ab + (long)r * aRowStride + kt + BK + ldk, 0, 1);
      int c = n0 + ldr;
      if (c < N) __builtin_prefetch(Bb + (long)c * Kd + kt + BK + ldk, 0, 1);
    }

    // ---- branch-free fragment reads from LDS + 2 WMMAs ----
    v16h a, b0, b1;
    {
      v8h_t lo = *(const v8h_t*)&As[sm * 16 + fr][kb];
      v8h_t hi = *(const v8h_t*)&As[sm * 16 + fr][kb + 16];
#pragma unroll
      for (int j = 0; j < 8; ++j) { a[j] = lo[j]; a[j + 8] = hi[j]; }
    }
    {
      v8h_t lo = *(const v8h_t*)&Bs[sn0 * 16 + fr][kb];
      v8h_t hi = *(const v8h_t*)&Bs[sn0 * 16 + fr][kb + 16];
#pragma unroll
      for (int j = 0; j < 8; ++j) { b0[j] = lo[j]; b0[j + 8] = hi[j]; }
    }
    {
      v8h_t lo = *(const v8h_t*)&Bs[(sn0 + 1) * 16 + fr][kb];
      v8h_t hi = *(const v8h_t*)&Bs[(sn0 + 1) * 16 + fr][kb + 16];
#pragma unroll
      for (int j = 0; j < 8; ++j) { b1[j] = lo[j]; b1[j + 8] = hi[j]; }
    }
    acc0 = __builtin_amdgcn_wmma_f32_16x16x32_f16(false, a, false, b0,
                                                  (short)0, acc0, false, false);
    acc1 = __builtin_amdgcn_wmma_f32_16x16x32_f16(false, a, false, b1,
                                                  (short)0, acc1, false, false);
    __syncthreads();
  }

  // ---- epilogue.  C layout: lane&15 = column, VGPR r -> row r + 8*(lane>>4)
  int halfSel = lane >> 4;
  if (act == 2) {
#pragma unroll
    for (int t = 0; t < 2; ++t) {
      v8f acc = t ? acc1 : acc0;
      int p = n0 + (sn0 + t) * 16 + fr;
#pragma unroll
      for (int r = 0; r < 8; ++r) {
        int m = m0 + sm * 16 + r + 8 * halfSel;
        float prod = (m < M && p < N)
                       ? acc[r] * (float)tsFused[((long)m * KK + bz) * DD + p]
                       : 0.f;
#pragma unroll
        for (int off = 8; off >= 1; off >>= 1) prod += __shfl_xor(prod, off, 16);
        if ((lane & 15) == 0 && m < M)
          atomicAdd(&logitsFused[m * KK + bz], prod);
      }
    }
  } else {
#pragma unroll
    for (int t = 0; t < 2; ++t) {
      v8f acc = t ? acc1 : acc0;
      int n = n0 + (sn0 + t) * 16 + fr;
      if (n >= N) continue;
      float bv = bias ? bias[n] : 0.f;
#pragma unroll
      for (int r = 0; r < 8; ++r) {
        int m = m0 + sm * 16 + r + 8 * halfSel;
        if (m < M) {
          float v = acc[r] + bv;
          if (act == 1) Ch[(long)bz * cBatch + (long)m * N + n] = (_Float16)tanhf(v);
          else          Cf[(long)bz * cBatch + (long)m * N + n] = v;
        }
      }
    }
  }
}

// --------------------------- helper kernels --------------------------------

__global__ void cast_f16_kernel(const float* __restrict__ in,
                                _Float16* __restrict__ out, long n) {
  long i = (long)blockIdx.x * blockDim.x + threadIdx.x;
  if (i < n) out[i] = (_Float16)in[i];
}

// out[b][c][r] = (f16) in[b][r][c]   (build column-major f16 weights)
__global__ void transpose_f16_kernel(const float* __restrict__ in,
                                     _Float16* __restrict__ out, int R_, int C_) {
  long b = blockIdx.y;
  long i = (long)blockIdx.x * blockDim.x + threadIdx.x;
  long n = (long)R_ * C_;
  if (i < n) {
    int r = (int)(i / C_), c = (int)(i % C_);
    out[b * n + (long)c * R_ + r] = (_Float16)in[b * n + i];
  }
}

// emb[b,e,m,:] = seq[b, clip(pos+1), :] * mask  (f32 + f16 copies)
__global__ void gather_emb_kernel(const float* __restrict__ seq,
                                  const int* __restrict__ mpos,
                                  const float* __restrict__ mask,
                                  float* __restrict__ emb,
                                  _Float16* __restrict__ emb16) {
  int idx = blockIdx.x;                    // (b*NE+e)*NM+m
  int pos = mpos[idx] + 1;
  pos = min(max(pos, 0), LL - 1);
  float mk = mask[idx];
  int b = idx / (NE * NM);
  const float* src = seq + ((long)b * LL + pos) * DD;
  long o = (long)idx * DD;
  for (int d = threadIdx.x; d < DD; d += blockDim.x) {
    float v = src[d] * mk;
    emb[o + d] = v;
    emb16[o + d] = (_Float16)v;
  }
}

// entity_as[b,e,h,l] = sum_m att[b,h,pos,l]*mask / max(cnt,1)
__global__ void entity_as_kernel(const float* __restrict__ att,
                                 const int* __restrict__ mpos,
                                 const float* __restrict__ mask,
                                 float* __restrict__ eas) {
  int idx = blockIdx.x;                    // (b*NE+e)*HH + h
  int hh = idx % HH;
  int be = idx / HH;                       // b*NE+e
  int b = be / NE;
  int pos[NM]; float mk[NM]; float cnt = 0.f;
#pragma unroll
  for (int m = 0; m < NM; ++m) {
    mk[m] = mask[be * NM + m];
    cnt += mk[m];
    int p = mpos[be * NM + m] + 1;
    pos[m] = min(max(p, 0), LL - 1);
  }
  float inv = 1.f / fmaxf(cnt, 1.f);
  const float* Ab = att + ((long)b * HH + hh) * LL * LL;
  float* dst = eas + (long)idx * LL;
  for (int l = threadIdx.x; l < LL; l += blockDim.x) {
    float s = 0.f;
#pragma unroll
    for (int m = 0; m < NM; ++m) s += Ab[(long)pos[m] * LL + l] * mk[m];
    dst[l] = s * inv;
  }
}

// masked softmax over mentions (axis m), in place on scores [b,e,m,K]
__global__ void softmax_kernel(float* __restrict__ scores,
                               const float* __restrict__ mask) {
  int t = blockIdx.x * blockDim.x + threadIdx.x;   // (b*NE+e)*KK + k
  if (t >= BS * NE * KK) return;
  int k = t % KK;
  int be = t / KK;
  float v[NM]; float mx = -1e30f;
#pragma unroll
  for (int m = 0; m < NM; ++m) {
    float s = scores[((long)be * NM + m) * KK + k];
    if (mask[be * NM + m] == 0.f) s += -1e6f;
    v[m] = s; mx = fmaxf(mx, s);
  }
  float sum = 0.f;
#pragma unroll
  for (int m = 0; m < NM; ++m) { v[m] = __expf(v[m] - mx); sum += v[m]; }
  float inv = 1.f / sum;
#pragma unroll
  for (int m = 0; m < NM; ++m)
    scores[((long)be * NM + m) * KK + k] = v[m] * inv;
}

// entity_es[b,e,k,:] = sum_m w[b,e,m,k] * emb[b,e,m,:]   (f16 out)
__global__ void entity_es_kernel(const float* __restrict__ w,
                                 const float* __restrict__ emb,
                                 _Float16* __restrict__ es) {
  int idx = blockIdx.x;                    // (b*NE+e)*KK + k
  int k = idx % KK;
  int be = idx / KK;
  float wm[NM];
#pragma unroll
  for (int m = 0; m < NM; ++m) wm[m] = w[((long)be * NM + m) * KK + k];
  const float* E = emb + (long)be * NM * DD;
  _Float16* dst = es + (long)idx * DD;
  for (int d = threadIdx.x; d < DD; d += blockDim.x) {
    float s = 0.f;
#pragma unroll
    for (int m = 0; m < NM; ++m) s += wm[m] * E[m * DD + d];
    dst[d] = (_Float16)s;
  }
}

// ht[b,i,j,l] = (1/h) * sum_h eas[b,i,h,l]*eas[b,j,h,l]
__global__ void ht_pair_kernel(const float* __restrict__ eas,
                               float* __restrict__ ht) {
  int l = blockIdx.x, b = blockIdx.y;
  __shared__ float s[NE * HH];
  for (int i = threadIdx.x; i < NE * HH; i += blockDim.x)
    s[i] = eas[((long)b * NE * HH + i) * LL + l];
  __syncthreads();
  for (int p = threadIdx.x; p < NE * NE; p += blockDim.x) {
    int i = p / NE, j = p % NE;
    float acc = 0.f;
#pragma unroll
    for (int hh = 0; hh < HH; ++hh) acc += s[i * HH + hh] * s[j * HH + hh];
    ht[(((long)b * NE + i) * NE + j) * LL + l] = acc * (1.f / (float)HH);
  }
}

// normalize ht over l, emit f16
__global__ void ht_norm_kernel(const float* __restrict__ ht,
                               _Float16* __restrict__ ht16) {
  long row = blockIdx.x;
  const float* p = ht + row * LL;
  float s = 0.f;
  for (int l = threadIdx.x; l < LL; l += 256) s += p[l];
  __shared__ float red[256];
  red[threadIdx.x] = s;
  __syncthreads();
  for (int off = 128; off > 0; off >>= 1) {
    if (threadIdx.x < off) red[threadIdx.x] += red[threadIdx.x + off];
    __syncthreads();
  }
  float inv = 1.f / (red[0] + 1e-5f);
  for (int l = threadIdx.x; l < LL; l += 256)
    ht16[row * LL + l] = (_Float16)(p[l] * inv);
}

// x[cell,c] = fmap[cell,:] @ Wlin[:,c] + blin[c]
__global__ void lin_kernel(const float* __restrict__ fmap,
                           const float* __restrict__ Wlin,
                           const float* __restrict__ blin,
                           float* __restrict__ x) {
  int o = blockIdx.x * blockDim.x + threadIdx.x;
  if (o >= BS * NE * NE * 3) return;
  int c = o % 3; long cell = o / 3;
  const float* fp = fmap + cell * DD;
  float s = blin[c];
  for (int d = 0; d < DD; ++d) s += fp[d] * Wlin[d * 3 + c];
  x[cell * 3 + c] = s;
}

// 3x3 SAME conv NHWC * HWIO + relu
__global__ void conv_kernel(const float* __restrict__ x,
                            const float* __restrict__ Wseg,
                            const float* __restrict__ bseg,
                            float* __restrict__ amap) {
  int cell = blockIdx.x;                   // (b*NE+i)*NE + j
  int j = cell % NE, i = (cell / NE) % NE, b = cell / (NE * NE);
  __shared__ float xs[27];
  if (threadIdx.x < 27) {
    int c = threadIdx.x % 3, dx = (threadIdx.x / 3) % 3, dy = threadIdx.x / 9;
    int ii = i + dy - 1, jj = j + dx - 1;
    xs[threadIdx.x] = (ii >= 0 && ii < NE && jj >= 0 && jj < NE)
                        ? x[(((long)b * NE + ii) * NE + jj) * 3 + c] : 0.f;
  }
  __syncthreads();
  for (int f = threadIdx.x; f < FF; f += blockDim.x) {
    float s = bseg[f];
#pragma unroll
    for (int t = 0; t < 27; ++t) s += xs[t] * Wseg[t * FF + f];
    amap[(long)cell * FF + f] = fmaxf(s, 0.f);
  }
}

// assemble Xh/Xt rows: [ es[b,hi/ti,k,:] | f16(amap[b,hi,ti,:]) ]
__global__ void buildX_kernel(const _Float16* __restrict__ es,
                              const float* __restrict__ amap,
                              const int* __restrict__ pairs,
                              _Float16* __restrict__ Xh,
                              _Float16* __restrict__ Xt) {
  long row = blockIdx.x;                   // n*KK + k
  int k = (int)(row % KK);
  long n = row / KK;
  int p = (int)(n & (PP - 1));
  int b = (int)(n >> 7);
  int hi = pairs[((long)b * PP + p) * 2 + 0];
  int ti = pairs[((long)b * PP + p) * 2 + 1];
  const _Float16* eh = es + (((long)(b * NE + hi)) * KK + k) * DD;
  const _Float16* et = es + (((long)(b * NE + ti)) * KK + k) * DD;
  const float* am = amap + (((long)b * NE + hi) * NE + ti) * FF;
  for (int c = threadIdx.x; c < CAT; c += blockDim.x) {
    _Float16 hv, tv;
    if (c < DD) { hv = eh[c]; tv = et[c]; }
    else        { _Float16 a = (_Float16)am[c - DD]; hv = a; tv = a; }
    Xh[row * CAT + c] = hv;
    Xt[row * CAT + c] = tv;
  }
}

__global__ void init_logits_kernel(float* __restrict__ logits,
                                   const float* __restrict__ bias) {
  int t = blockIdx.x * blockDim.x + threadIdx.x;
  if (t < NROW) logits[t] = bias[t % KK];
}

// ---------------------------------------------------------------------------
static inline char* ws_take(char*& p, size_t bytes) {
  char* r = p;
  p += (bytes + 255) & ~(size_t)255;
  return r;
}

extern "C" void kernel_launch(void* const* d_in, const int* in_sizes, int n_in,
                              void* d_out, int out_size, void* d_ws, size_t ws_size,
                              hipStream_t stream) {
  (void)in_sizes; (void)n_in; (void)out_size; (void)ws_size;
  const float* seq      = (const float*)d_in[0];
  const float* att      = (const float*)d_in[1];
  const float* mask     = (const float*)d_in[2];
  const int*   mpos     = (const int*)d_in[3];
  const int*   pairs    = (const int*)d_in[4];
  const float* Wattn    = (const float*)d_in[5];
  const float* battn    = (const float*)d_in[6];
  const float* attn_net = (const float*)d_in[7];
  const float* Wlin     = (const float*)d_in[8];
  const float* blin     = (const float*)d_in[9];
  const float* Wseg     = (const float*)d_in[10];
  const float* bseg     = (const float*)d_in[11];
  const float* Whead    = (const float*)d_in[12];
  const float* bhead    = (const float*)d_in[13];
  const float* Wtail    = (const float*)d_in[14];
  const float* btail    = (const float*)d_in[15];
  const float* bil      = (const float*)d_in[16];
  const float* bilb     = (const float*)d_in[17];
  float* logits = (float*)d_out;

  // ---- workspace layout -------------------------------------------------
  char* wp = (char*)d_ws;
  float*    emb     = (float*)   ws_take(wp, (size_t)BS*NE*NM*DD*4);
  _Float16* emb16   = (_Float16*)ws_take(wp, (size_t)BS*NE*NM*DD*2);
  float*    eas     = (float*)   ws_take(wp, (size_t)BS*NE*HH*LL*4);
  _Float16* Wattncm = (_Float16*)ws_take(wp, (size_t)DD*RR*2);
  _Float16* anet16  = (_Float16*)ws_take(wp, (size_t)KK*RR*2);
  _Float16* Hth16   = (_Float16*)ws_take(wp, (size_t)BS*NE*NM*RR*2);
  float*    scores  = (float*)   ws_take(wp, (size_t)BS*NE*NM*KK*4);
  _Float16* es16    = (_Float16*)ws_take(wp, (size_t)BS*NE*KK*DD*2);
  float*    htf     = (float*)   ws_take(wp, (size_t)BS*NE*NE*LL*4);
  _Float16* ht16    = (_Float16*)ws_take(wp, (size_t)BS*NE*NE*LL*2);
  _Float16* seqcm   = (_Float16*)ws_take(wp, (size_t)BS*DD*LL*2);
  float*    fmap    = (float*)   ws_take(wp, (size_t)BS*NE*NE*DD*4);
  float*    xlin    = (float*)   ws_take(wp, (size_t)BS*NE*NE*3*4);
  float*    amap    = (float*)   ws_take(wp, (size_t)BS*NE*NE*FF*4);
  _Float16* Whcm    = (_Float16*)ws_take(wp, (size_t)CAT*DD*2);
  _Float16* Wtcm    = (_Float16*)ws_take(wp, (size_t)CAT*DD*2);
  _Float16* bilcm   = (_Float16*)ws_take(wp, (size_t)KK*DD*DD*2);
  _Float16* Xh      = (_Float16*)ws_take(wp, (size_t)NROW*CAT*2);
  _Float16* Xt      = (_Float16*)ws_take(wp, (size_t)NROW*CAT*2);
  _Float16* hs16    = (_Float16*)ws_take(wp, (size_t)NROW*DD*2);
  _Float16* ts16    = (_Float16*)ws_take(wp, (size_t)NROW*DD*2);

  // ---- weight conversion (f32 -> col-major f16) -------------------------
  {
    long n = (long)DD * RR;
    transpose_f16_kernel<<<dim3((n + 255) / 256, 1), 256, 0, stream>>>(Wattn, Wattncm, DD, RR);
    n = (long)KK * RR;          // attn_net.T col-major == attn_net row-major
    cast_f16_kernel<<<(n + 255) / 256, 256, 0, stream>>>(attn_net, anet16, n);
    n = (long)CAT * DD;
    transpose_f16_kernel<<<dim3((n + 255) / 256, 1), 256, 0, stream>>>(Whead, Whcm, CAT, DD);
    transpose_f16_kernel<<<dim3((n + 255) / 256, 1), 256, 0, stream>>>(Wtail, Wtcm, CAT, DD);
    n = (long)LL * DD;
    transpose_f16_kernel<<<dim3((n + 255) / 256, BS), 256, 0, stream>>>(seq, seqcm, LL, DD);
    n = (long)DD * DD;
    transpose_f16_kernel<<<dim3((n + 255) / 256, KK), 256, 0, stream>>>(bil, bilcm, DD, DD);
  }

  // ---- mention gathers ---------------------------------------------------
  gather_emb_kernel<<<BS * NE * NM, 256, 0, stream>>>(seq, mpos, mask, emb, emb16);
  entity_as_kernel<<<BS * NE * HH, 256, 0, stream>>>(att, mpos, mask, eas);

  // ---- attention-pooling MLP: tanh(emb@Wattn+battn) @ attn_net.T --------
  wmma_gemm_block<<<dim3(RR / BN, (BS * NE * NM + BM - 1) / BM, 1), 256, 0, stream>>>(
      emb16, DD, 0, Wattncm, 0, battn, nullptr, Hth16, 0, nullptr, nullptr,
      BS * NE * NM, RR, DD, 1);
  wmma_gemm_block<<<dim3((KK + BN - 1) / BN, (BS * NE * NM + BM - 1) / BM, 1), 256, 0, stream>>>(
      Hth16, RR, 0, anet16, 0, nullptr, scores, nullptr, 0, nullptr, nullptr,
      BS * NE * NM, KK, RR, 0);
  softmax_kernel<<<(BS * NE * KK + 255) / 256, 256, 0, stream>>>(scores, mask);
  entity_es_kernel<<<BS * NE * KK, 256, 0, stream>>>(scores, emb, es16);

  // ---- pairwise attention overlap map -----------------------------------
  ht_pair_kernel<<<dim3(LL, BS), 256, 0, stream>>>(eas, htf);
  ht_norm_kernel<<<BS * NE * NE, 256, 0, stream>>>(htf, ht16);
  wmma_gemm_block<<<dim3(DD / BN, (NE * NE + BM - 1) / BM, BS), 256, 0, stream>>>(
      ht16, LL, (long)NE * NE * LL, seqcm, (long)DD * LL, nullptr,
      fmap, nullptr, (long)NE * NE * DD, nullptr, nullptr, NE * NE, DD, LL, 0);
  lin_kernel<<<(BS * NE * NE * 3 + 255) / 256, 256, 0, stream>>>(fmap, Wlin, blin, xlin);
  conv_kernel<<<BS * NE * NE, 256, 0, stream>>>(xlin, Wseg, bseg, amap);

  // ---- head/tail projections (dominant GEMMs) ---------------------------
  buildX_kernel<<<NROW, 256, 0, stream>>>(es16, amap, pairs, Xh, Xt);
  wmma_gemm_block<<<dim3(DD / BN, NROW / BM, 1), 256, 0, stream>>>(
      Xh, CAT, 0, Whcm, 0, bhead, nullptr, hs16, 0, nullptr, nullptr,
      NROW, DD, CAT, 1);
  wmma_gemm_block<<<dim3(DD / BN, NROW / BM, 1), 256, 0, stream>>>(
      Xt, CAT, 0, Wtcm, 0, btail, nullptr, ts16, 0, nullptr, nullptr,
      NROW, DD, CAT, 1);

  // ---- bilinear classifier (fused dot epilogue, batched over k) ---------
  init_logits_kernel<<<(NROW + 255) / 256, 256, 0, stream>>>(logits, bilb);
  wmma_gemm_block<<<dim3(DD / BN, NPAIR / BM, KK), 256, 0, stream>>>(
      hs16, (long)KK * DD, DD, bilcm, (long)DD * DD, nullptr,
      nullptr, nullptr, 0, ts16, logits, NPAIR, DD, DD, 2);
}
